// ImplicitTopology_54039278518512
// MI455X (gfx1250) — matrix operand order repaired and checked by reading, not compiled, not run on hardware
//
#include <hip/hip_runtime.h>
#include <hip/hip_bf16.h>

typedef __attribute__((ext_vector_type(16))) _Float16 v16h;
typedef __attribute__((ext_vector_type(8)))  _Float16 v8h;
typedef __attribute__((ext_vector_type(8)))  float    v8f;
typedef __attribute__((ext_vector_type(4)))  unsigned int u32x4;
typedef __attribute__((ext_vector_type(8)))  int      i32x8;
typedef __attribute__((ext_vector_type(4)))  int      i32x4;

#define BATCH 8
#define NSEQ  512
#define DDIM  128
#define HDIM  128

#if defined(__has_builtin)
#if __has_builtin(__builtin_amdgcn_tensor_load_to_lds) && \
    __has_builtin(__builtin_amdgcn_s_wait_tensorcnt)
#define HAVE_TDM 1
#endif
#endif

static __device__ __forceinline__ v16h cat16(v8h lo, v8h hi) {
    return __builtin_shufflevector(lo, hi, 0,1,2,3,4,5,6,7,8,9,10,11,12,13,14,15);
}

static __device__ __forceinline__ v16h relu16(v16h x) {
    const v16h z = (v16h)(_Float16)0.0f;
#if __has_builtin(__builtin_elementwise_maximumnum)
    return __builtin_elementwise_maximumnum(x, z);
#elif __has_builtin(__builtin_elementwise_maximum)
    return __builtin_elementwise_maximum(x, z);
#elif __has_builtin(__builtin_elementwise_max)
    return __builtin_elementwise_max(x, z);
#else
    #pragma unroll
    for (int i = 0; i < 16; ++i) x[i] = x[i] > (_Float16)0 ? x[i] : (_Float16)0;
    return x;
#endif
}

static __device__ __forceinline__ v8f wmma_f16(v16h a, v16h b, v8f c) {
    return __builtin_amdgcn_wmma_f32_16x16x32_f16(false, a, false, b, (short)0, c,
                                                  false, false);
}

#ifdef HAVE_TDM
// Issue one TDM transfer of a contiguous 4KB (2048 x 2B) region, global -> LDS.
// D# per ISA ch.8: group0 = {count=1 | lds_addr | global_addr(57b) | type=2},
// group1 = {wg_mask=0, data_size=1(2B), tensor_dim0=2048, tile_dim0=2048}.
// clang-23 toolchain: 6-arg builtin (g0, g1, g2, g3, extra, cpol).
static __device__ __forceinline__ void tdm_load_4kb(unsigned ldsOff, const void* gsrc) {
    unsigned long long ga = (unsigned long long)(uintptr_t)gsrc;
    u32x4 g0;
    g0[0] = 1u;                                   // count=1 (valid), user mode
    g0[1] = ldsOff;                               // lds_addr (bytes)
    g0[2] = (unsigned)(ga & 0xFFFFFFFFu);         // global_addr[31:0]
    g0[3] = (unsigned)((ga >> 32) & 0x01FFFFFFu)  // global_addr[56:32]
          | (2u << 30);                           // type = 2 ("image")
    i32x8 g1;
    g1[0] = 0x00010000;                           // data_size = 1 -> 2 bytes
    g1[1] = (int)(2048u << 16);                   // tensor_dim0[15:0] = 2048
    g1[2] = (int)(1u << 16);                      // tensor_dim0 hi=0, tensor_dim1=1
    g1[3] = (int)(2048u << 16);                   // tile_dim0 = 2048
    g1[4] = 0;                                    // tile_dim1 = tile_dim2 = 0 (1D)
    g1[5] = 2048;                                 // tensor_dim0_stride (unused, 1D)
    g1[6] = 0;
    g1[7] = 0;
    i32x4 z4 = {0, 0, 0, 0};
    i32x8 z8 = {0, 0, 0, 0, 0, 0, 0, 0};
    __builtin_amdgcn_tensor_load_to_lds(g0, g1, z4, z4, z8, 0);
}
#endif

// ---------------------------------------------------------------------------
// Kernel 1: F0 (fp32) -> f16
// ---------------------------------------------------------------------------
__global__ void cvt_kernel(const float* __restrict__ src, _Float16* __restrict__ dst, int n) {
    int i = blockIdx.x * 256 + threadIdx.x;
    if (i < n) dst[i] = (_Float16)src[i];
}

// ---------------------------------------------------------------------------
// Kernel 2: pre-swizzle W matrices into CDNA5 16-bit B-operand layout.
// B operand (K=32 x N=16, f16): lane L holds col n = L%16; half slot s in
// [0,16) holds k = 8*(L/16) + s (s<8), 16 + 8*(L/16) + (s-8) (s>=8).  Buffer layout: [tile][lane][16 halves].
// ---------------------------------------------------------------------------
__global__ void pack_kernel(const float* __restrict__ W1, const float* __restrict__ W2,
                            _Float16* __restrict__ W2p,
                            _Float16* __restrict__ Wsp, _Float16* __restrict__ Wdp) {
    int t = threadIdx.x;
    // W2 pack: K = H = 128 (4 chunks of 32), N = 64 (4 tiles of 16). 8192 halves.
    for (int e = t; e < 8192; e += 256) {
        int s = e & 15, lane = (e >> 4) & 31, nt = (e >> 9) & 3, kc = e >> 11;
        int k = ((s >> 3) << 4) + ((lane >> 4) << 3) + (s & 7);
        int n = lane & 15;
        int h = kc * 32 + k;        // K index = hidden channel
        int o = nt * 16 + n;        // N index = output channel
        W2p[e] = (_Float16)W2[o * HDIM + h];
    }
    // Wsum/Wdiff pack: K = D = 128 (4 chunks), N = H = 128 (8 tiles). 16384 halves.
    for (int e = t; e < 16384; e += 256) {
        int s = e & 15, lane = (e >> 4) & 31, nt = (e >> 9) & 7, kc = e >> 12;
        int kd = kc * 32 + ((s >> 3) << 4) + ((lane >> 4) << 3) + (s & 7);   // K index = d
        int hh = nt * 16 + (lane & 15);              // N index = h
        float a = W1[hh * 384 + kd];
        float b = W1[hh * 384 + 128 + kd];
        float c = W1[hh * 384 + 256 + kd];
        Wsp[e] = (_Float16)(a + c);
        Wdp[e] = (_Float16)(b - c);
    }
}

// ---------------------------------------------------------------------------
// Kernel 3: P = F0 @ Wsum^T + b1,  Q = F0 @ Wdiff^T   (f16 out, row-major)
// One wave per 16-row block of F0 (4096/16 = 256 blocks).
// ---------------------------------------------------------------------------
__global__ __launch_bounds__(32)
void pq_kernel(const _Float16* __restrict__ F0h,
               const _Float16* __restrict__ Wsp, const _Float16* __restrict__ Wdp,
               const float* __restrict__ b1,
               _Float16* __restrict__ P, _Float16* __restrict__ Q) {
    int rb   = blockIdx.x;          // row block (16 rows)
    int lane = threadIdx.x;
    int m    = lane & 15;
    int g    = lane >> 4;

    __shared__ v8h sA[256];         // 16 rows x 128 halves
    const v8h* gA = (const v8h*)(F0h + (size_t)rb * 16 * DDIM);
#ifdef HAVE_TDM
    tdm_load_4kb((unsigned)(uintptr_t)&sA[0], (const void*)gA);
    __builtin_amdgcn_s_wait_tensorcnt((short)0);
#else
    #pragma unroll
    for (int t = 0; t < 8; ++t) sA[t * 32 + lane] = gA[t * 32 + lane];
#endif
    __syncthreads();

    // A operand (16x32 f16): lane L = row m; halves 0..7 -> k = 8*g + s,
    // halves 8..15 -> k = 16 + 8*g + (s-8).  Two b128 LDS loads per chunk.
    v16h aF[4];
    #pragma unroll
    for (int kc = 0; kc < 4; ++kc) {
        int idx = m * 16 + kc * 4 + g;
        aF[kc] = cat16(sA[idx], sA[idx + 2]);
    }

    const v8h* gBs = (const v8h*)Wsp;
    const v8h* gBd = (const v8h*)Wdp;

    #pragma unroll
    for (int nt = 0; nt < 8; ++nt) {
        v8f ap = (v8f)0.0f, aq = (v8f)0.0f;
        #pragma unroll
        for (int kc = 0; kc < 4; ++kc) {
            int e = ((kc * 8 + nt) * 32 + lane) * 2;
            v16h bs = cat16(gBs[e], gBs[e + 1]);
            v16h bd = cat16(gBd[e], gBd[e + 1]);
            ap = wmma_f16(aF[kc], bs, ap);
            aq = wmma_f16(aF[kc], bd, aq);
        }
        // D layout: lane L holds col n = L%16; VGPR v holds row m = v + 8*(L/16)
        float bias = b1[nt * 16 + m];
        #pragma unroll
        for (int v = 0; v < 8; ++v) {
            int row = rb * 16 + v + 8 * g;
            P[(size_t)row * HDIM + nt * 16 + m] = (_Float16)(ap[v] + bias);
            Q[(size_t)row * HDIM + nt * 16 + m] = (_Float16)(aq[v]);
        }
    }
}

// ---------------------------------------------------------------------------
// Kernel 4: pairwise fused MLP. One wave per 16x16 (i,j) tile.
//   Tiles staged to LDS by the Tensor Data Mover; W2 B-operands fetched into
//   VGPRs while the DMA runs; s_wait_tensorcnt before A-operand gather.
// ---------------------------------------------------------------------------
__global__ __launch_bounds__(32)
void pair_kernel(const _Float16* __restrict__ P, const _Float16* __restrict__ Q,
                 const _Float16* __restrict__ W2p,
                 const float* __restrict__ b2, const float* __restrict__ W3,
                 const float* __restrict__ b3, const int* __restrict__ mask,
                 float* __restrict__ out) {
    int it = blockIdx.x, jt = blockIdx.y, b = blockIdx.z;
    int i0 = it * 16, j0 = jt * 16;
    int lane = threadIdx.x;
    int m    = lane & 15;
    int g    = lane >> 4;

    __shared__ v8h  sP[256];        // 16 x 128 halves
    __shared__ v8h  sQ[256];
    __shared__ float sMaskI[16];

    const v8h* gP = (const v8h*)(P + (size_t)(b * NSEQ + i0) * HDIM);
    const v8h* gQ = (const v8h*)(Q + (size_t)(b * NSEQ + j0) * HDIM);

#ifdef HAVE_TDM
    tdm_load_4kb((unsigned)(uintptr_t)&sP[0], (const void*)gP);
    tdm_load_4kb((unsigned)(uintptr_t)&sQ[0], (const void*)gQ);
#else
    #pragma unroll
    for (int t = 0; t < 8; ++t) {
        sP[t * 32 + lane] = gP[t * 32 + lane];
        sQ[t * 32 + lane] = gQ[t * 32 + lane];
    }
#endif
    if (lane < 16) sMaskI[lane] = mask[b * NSEQ + i0 + lane] ? 0.0f : -2.0e9f;

    // Overlap with the TDM transfer: W2 B-operands resident in VGPRs
    // (16 tiles x 8 VGPRs), plus per-lane W3/b2 slices.
    const v8h* gB = (const v8h*)W2p;
    v16h Bw[4][4];
    #pragma unroll
    for (int kc = 0; kc < 4; ++kc)
        #pragma unroll
        for (int nt = 0; nt < 4; ++nt) {
            int e = ((kc * 4 + nt) * 32 + lane) * 2;
            Bw[kc][nt] = cat16(gB[e], gB[e + 1]);
        }

    float w3v[4], b2v[4];
    #pragma unroll
    for (int nt = 0; nt < 4; ++nt) {
        w3v[nt] = W3[nt * 16 + m];
        b2v[nt] = b2[nt * 16 + m];
    }
    float b3v = b3[0];

#ifdef HAVE_TDM
    __builtin_amdgcn_s_wait_tensorcnt((short)0);
#endif
    __syncthreads();

    // Pre-gather P in A-operand layout (invariant over j)
    v16h aP[4];
    #pragma unroll
    for (int kc = 0; kc < 4; ++kc) {
        int idx = m * 16 + kc * 4 + g;
        aP[kc] = cat16(sP[idx], sP[idx + 2]);
    }

    for (int j = 0; j < 16; ++j) {
        v8f acc[4];
        #pragma unroll
        for (int nt = 0; nt < 4; ++nt) acc[nt] = (v8f)0.0f;

        #pragma unroll
        for (int kc = 0; kc < 4; ++kc) {
            int idx = j * 16 + kc * 4 + g;        // Q row j, broadcast per 16-group
            v16h q  = cat16(sQ[idx], sQ[idx + 2]);
            v16h h1 = relu16(aP[kc] + q);
            #pragma unroll
            for (int nt = 0; nt < 4; ++nt)
                acc[nt] = wmma_f16(h1, Bw[kc][nt], acc[nt]);
        }

        // Epilogue: part[v] = sum_o W3[o]*relu(acc+b2) ; o striped over lanes
        float part[8];
        #pragma unroll
        for (int v = 0; v < 8; ++v) part[v] = 0.0f;
        #pragma unroll
        for (int nt = 0; nt < 4; ++nt)
            #pragma unroll
            for (int v = 0; v < 8; ++v)
                part[v] += fmaxf(acc[nt][v] + b2v[nt], 0.0f) * w3v[nt];

        #pragma unroll
        for (int x = 1; x < 16; x <<= 1)
            #pragma unroll
            for (int v = 0; v < 8; ++v)
                part[v] += __shfl_xor(part[v], x, 32);

        float jadd = mask[b * NSEQ + j0 + j] ? 0.0f : -2.0e9f;

        if ((lane & 15) == 0) {
            int mrow = g * 8;   // lane 0 -> rows 0..7, lane 16 -> rows 8..15
            #pragma unroll
            for (int v = 0; v < 8; ++v) {
                float lg = part[v] + b3v + sMaskI[mrow + v] + jadd;
                float sg = 1.0f / (1.0f + __expf(-lg));
                out[((size_t)(b * NSEQ) + (i0 + mrow + v)) * NSEQ + (j0 + j)] = sg;
            }
        }
    }
}

// ---------------------------------------------------------------------------
extern "C" void kernel_launch(void* const* d_in, const int* in_sizes, int n_in,
                              void* d_out, int out_size, void* d_ws, size_t ws_size,
                              hipStream_t stream) {
    const float* F0   = (const float*)d_in[0];
    const int*   mask = (const int*)d_in[1];
    const float* W1   = (const float*)d_in[2];
    const float* b1   = (const float*)d_in[3];
    const float* W2   = (const float*)d_in[4];
    const float* b2   = (const float*)d_in[5];
    const float* W3   = (const float*)d_in[6];
    const float* b3   = (const float*)d_in[7];

    char* ws = (char*)d_ws;
    _Float16* F0h = (_Float16*)(ws);                               // 1 MB
    _Float16* P   = (_Float16*)(ws + (1 << 20));                   // 1 MB
    _Float16* Q   = (_Float16*)(ws + (2 << 20));                   // 1 MB
    _Float16* W2p = (_Float16*)(ws + (3 << 20));                   // 16 KB
    _Float16* Wsp = (_Float16*)(ws + (3 << 20) + (16 << 10));      // 32 KB
    _Float16* Wdp = (_Float16*)(ws + (3 << 20) + (48 << 10));      // 32 KB

    const int nF0 = BATCH * NSEQ * DDIM;
    cvt_kernel<<<(nF0 + 255) / 256, 256, 0, stream>>>(F0, F0h, nF0);
    pack_kernel<<<1, 256, 0, stream>>>(W1, W2, W2p, Wsp, Wdp);
    pq_kernel<<<(BATCH * NSEQ) / 16, 32, 0, stream>>>(F0h, Wsp, Wdp, b1, P, Q);

    dim3 grid(NSEQ / 16, NSEQ / 16, BATCH);
    pair_kernel<<<grid, 32, 0, stream>>>(P, Q, W2p, b2, W3, b3, mask, (float*)d_out);
}
